// GraphSage_14482629722967
// MI455X (gfx1250) — matrix operand order repaired and checked by reading, not compiled, not run on hardware
//
#include <hip/hip_runtime.h>
#include <hip/hip_bf16.h>

// ---------------------------------------------------------------------------
// GraphSAGE forward, MI455X (gfx1250).
//   * all GEMMs: bf16 WMMA (v_wmma_f32_16x16x32_bf16), f32 accumulate
//   * feature matrix converted to bf16 ONCE (51 MB, L2-resident) -> halves the
//     dominant hop-2 gather traffic (840 MB -> 420 MB out of L2)
//   * GEMM A tiles staged with GLOBAL_LOAD_ASYNC_TO_LDS_B128 (ASYNCcnt DMA)
//   * intermediates (x1/x2/h1_full/feat0_h/h1_mean) kept in bf16; output f32
// F=256, H1=H2=256, K=2F=512. All M dims multiples of 16.
// ---------------------------------------------------------------------------

typedef __attribute__((ext_vector_type(16))) __bf16 bf16x16;
typedef __attribute__((ext_vector_type(8)))  __bf16 bf16x8;
typedef __attribute__((ext_vector_type(8)))  float  f32x8;

#define FEAT 256
#define KDIM 512   // 2*FEAT
#define KT   16    // K tiles of 32 -> 512

// ---------------------------------------------------------------------------
// fp32 -> bf16 feature matrix conversion (one pass; 8 elements / thread)
// ---------------------------------------------------------------------------
__global__ void sage_cvt_bf16(const float* __restrict__ src,
                              __bf16* __restrict__ dst, int n8) {
    int t = blockIdx.x * blockDim.x + threadIdx.x;
    if (t >= n8) return;
    const float4 f0 = ((const float4*)src)[t * 2 + 0];
    const float4 f1 = ((const float4*)src)[t * 2 + 1];
    bf16x8 p;
    p[0] = (__bf16)f0.x; p[1] = (__bf16)f0.y;
    p[2] = (__bf16)f0.z; p[3] = (__bf16)f0.w;
    p[4] = (__bf16)f1.x; p[5] = (__bf16)f1.y;
    p[6] = (__bf16)f1.z; p[7] = (__bf16)f1.w;
    *(bf16x8*)(dst + (size_t)t * 8) = p;
}

// ---------------------------------------------------------------------------
// Pack fp32 weights W[k][n] (K=512, N=256, row-major) into per-lane bf16
// B fragments: t = ((nt*KT + kt)*32 + lane)*16 + j
//   lane encodes N = nt*16 + (lane&15), K half = (lane&16)
//   element j -> K = kt*32 + (lane&16)/2 + (j&7) + (j&8)*2   (16-bit B VGPR layout)
// ---------------------------------------------------------------------------
__global__ void sage_pack_w(const float* __restrict__ W, __bf16* __restrict__ wp) {
    int t = blockIdx.x * blockDim.x + threadIdx.x;   // 131072 threads total
    int j    =  t        & 15;
    int lane = (t >> 4)  & 31;
    int kt   = (t >> 9)  & 15;
    int nt   =  t >> 13;
    int k = kt * 32 + ((lane & 16) >> 1) + (j & 7) + ((j & 8) << 1);
    int n = nt * 16 + (lane & 15);
    wp[t] = (__bf16)W[(size_t)k * FEAT + n];
}

// ---------------------------------------------------------------------------
// out[r][:] = mean_s featb[idx[r*S+s]][:]   (bf16 in/out, f32 accumulate)
// 32 threads per row, 8 columns (one bf16x8 = 16B) per thread.
// ---------------------------------------------------------------------------
__global__ void sage_gather_mean(const __bf16* __restrict__ featb,
                                 const int* __restrict__ idx,
                                 __bf16* __restrict__ out, int R, int S) {
    int t  = blockIdx.x * blockDim.x + threadIdx.x;
    int r  = t >> 5;
    int c8 = (t & 31) << 3;
    if (r >= R) return;
    const int* row = idx + (size_t)r * S;
    float acc[8];
#pragma unroll
    for (int e = 0; e < 8; ++e) acc[e] = 0.f;
    for (int s = 0; s < S; ++s) {
        const bf16x8 v = *(const bf16x8*)(featb + (size_t)row[s] * FEAT + c8);
#pragma unroll
        for (int e = 0; e < 8; ++e) acc[e] += (float)v[e];
    }
    const float inv = 1.0f / (float)S;
    bf16x8 o;
#pragma unroll
    for (int e = 0; e < 8; ++e) o[e] = (__bf16)(acc[e] * inv);
    *(bf16x8*)(out + (size_t)r * FEAT + c8) = o;
}

// ---------------------------------------------------------------------------
// out[r][:] = mean over 10 consecutive rows of src (bf16 in/out, f32 acc)
// ---------------------------------------------------------------------------
__global__ void sage_group_mean10(const __bf16* __restrict__ src,
                                  __bf16* __restrict__ out, int R) {
    int t  = blockIdx.x * blockDim.x + threadIdx.x;
    int r  = t >> 5;
    int c8 = (t & 31) << 3;
    if (r >= R) return;
    float acc[8];
#pragma unroll
    for (int e = 0; e < 8; ++e) acc[e] = 0.f;
    for (int s = 0; s < 10; ++s) {
        const bf16x8 v = *(const bf16x8*)(src + (size_t)(r * 10 + s) * FEAT + c8);
#pragma unroll
        for (int e = 0; e < 8; ++e) acc[e] += (float)v[e];
    }
    bf16x8 o;
#pragma unroll
    for (int e = 0; e < 8; ++e) o[e] = (__bf16)(acc[e] * 0.1f);
    *(bf16x8*)(out + (size_t)r * FEAT + c8) = o;
}

// ---------------------------------------------------------------------------
// relu( [ a1[idx[m]] | a2[m] ] @ W ) -> outf (f32) or outb (bf16), Mx256, K=512
//   a1/a2: bf16 [.,256] sources (a1 gathered via idx; idx==nullptr -> direct)
//   wp   : packed bf16 weights (see sage_pack_w)
// Block = 512 threads = 16 waves; block owns a 16-row x 256-col output stripe.
// A tile staged via async global->LDS DMA: wave w stages row w with exactly
// two global_load_async_to_lds_b128 (gathered half + direct half), then
// s_wait_asynccnt 0 + workgroup barrier.
// ---------------------------------------------------------------------------
__global__ __launch_bounds__(512)
void sage_gemm_relu(const __bf16* __restrict__ a1, const int* __restrict__ idx,
                    const __bf16* __restrict__ a2, const __bf16* __restrict__ wp,
                    float* __restrict__ outf, __bf16* __restrict__ outb, int M) {
    // 8-element pad: row stride 1040 B -> the 16 rows of a fragment read land
    // on disjoint bank groups for ds_load_b128.
    __shared__ __align__(16) __bf16 As[16][KDIM + 8];

    const int m0   = blockIdx.x << 4;
    const int tid  = threadIdx.x;
    const int wave = tid >> 5;
    const int lane = tid & 31;

    // ---- stage A tile (16 x 512 bf16) via async DMA into LDS -------------
    {
        const int m = m0 + wave;                       // wave w stages row w
        const int g = idx ? idx[m] : m;
        // generic shared-memory pointer: low 32 bits == wave-relative LDS offset
        const unsigned ldsRow = (unsigned)(size_t)(&As[wave][0]);
        const unsigned la1 = ldsRow + (unsigned)(lane * 16);
        const unsigned la2 = la1 + 512;                // second half of the row
        const unsigned long long ga1 =
            (unsigned long long)(size_t)(a1 + (size_t)g * FEAT) + (unsigned)(lane * 16);
        const unsigned long long ga2 =
            (unsigned long long)(size_t)(a2 + (size_t)m * FEAT) + (unsigned)(lane * 16);
        asm volatile("global_load_async_to_lds_b128 %0, %1, off"
                     :: "v"(la1), "v"(ga1) : "memory");
        asm volatile("global_load_async_to_lds_b128 %0, %1, off"
                     :: "v"(la2), "v"(ga2) : "memory");
        asm volatile("s_wait_asynccnt 0" ::: "memory");
    }
    __syncthreads();

    // ---- WMMA main loop ---------------------------------------------------
    const int arow = lane & 15;                 // A: M = lane%16
    const int klo  = (lane < 16) ? 0 : 8;       // A: K-half select

    f32x8 acc = {0.f, 0.f, 0.f, 0.f, 0.f, 0.f, 0.f, 0.f};

#pragma unroll
    for (int kt = 0; kt < KT; ++kt) {
        // A fragment: two aligned 16B LDS reads (K klo..klo+7, K 16+klo..23+klo)
        const bf16x8 alo = *(const bf16x8*)&As[arow][kt * 32 + klo];
        const bf16x8 ahi = *(const bf16x8*)&As[arow][kt * 32 + 16 + klo];
        bf16x16 a;
#pragma unroll
        for (int j = 0; j < 8; ++j) { a[j] = alo[j]; a[j + 8] = ahi[j]; }

        // B fragment: contiguous 32 B per lane from packed weights
        const bf16x16 b =
            *(const bf16x16*)(wp + ((size_t)(wave * KT + kt) * 512 + lane * 16));

        acc = __builtin_amdgcn_wmma_f32_16x16x32_bf16(
            /*neg_a=*/false, a, /*neg_b=*/false, b,
            /*c_mod=*/(short)0, acc, /*reuse_a=*/false, /*reuse_b=*/false);
    }

    // ---- ReLU + store: VGPR r -> row (r or r+8), lane -> column ----------
    const int ncol  = (wave << 4) + (lane & 15);
    const int mbase = m0 + ((lane >> 4) << 3);
#pragma unroll
    for (int r = 0; r < 8; ++r) {
        float v = acc[r];
        v = v > 0.f ? v : 0.f;
        if (outb) outb[(size_t)(mbase + r) * FEAT + ncol] = (__bf16)v;
        else      outf[(size_t)(mbase + r) * FEAT + ncol] = v;
    }
}

// ---------------------------------------------------------------------------
extern "C" void kernel_launch(void* const* d_in, const int* in_sizes, int n_in,
                              void* d_out, int out_size, void* d_ws, size_t ws_size,
                              hipStream_t stream) {
    (void)in_sizes; (void)n_in; (void)out_size; (void)ws_size;

    const int*   forest0 = (const int*)d_in[0];   // [8192]
    const int*   forest1 = (const int*)d_in[1];   // [8192,10] (flat = 81920)
    const int*   forest2 = (const int*)d_in[2];   // [81920,10]
    const float* feat    = (const float*)d_in[3]; // [100000,256] f32
    const float* W1      = (const float*)d_in[4]; // [512,256]
    const float* W2      = (const float*)d_in[5]; // [512,256]
    float*       out     = (float*)d_out;         // [8192,256] f32

    const int NROOTS = 8192;
    const int NH1    = 81920;           // NROOTS * S1
    const int NNODES = 100000;

    char* ws = (char*)d_ws;
    size_t off = 0;
    auto carve = [&](size_t bytes) -> void* {
        void* p = ws + off;
        off = (off + bytes + 255) & ~(size_t)255;
        return p;
    };
    __bf16* featb = (__bf16*)carve((size_t)NNODES * FEAT * 2);  // bf16 features
    __bf16* x1b   = (__bf16*)carve((size_t)NROOTS * FEAT * 2);  // mean hop-1
    __bf16* x2b   = (__bf16*)carve((size_t)NH1    * FEAT * 2);  // mean hop-2
    __bf16* h1fb  = (__bf16*)carve((size_t)NH1    * FEAT * 2);  // h1 per hop-1 node
    __bf16* f0hb  = (__bf16*)carve((size_t)NROOTS * FEAT * 2);  // layer-1 root out
    __bf16* h1mb  = (__bf16*)carve((size_t)NROOTS * FEAT * 2);  // groupmean(h1f)
    __bf16* wp1   = (__bf16*)carve((size_t)KDIM * FEAT * 2);
    __bf16* wp2   = (__bf16*)carve((size_t)KDIM * FEAT * 2);

    // one-time conversions / packing
    const int n8 = NNODES * FEAT / 8;                            // 3.2M
    sage_cvt_bf16<<<(n8 + 255) / 256, 256, 0, stream>>>(feat, featb, n8);
    sage_pack_w<<<512, 256, 0, stream>>>(W1, wp1);
    sage_pack_w<<<512, 256, 0, stream>>>(W2, wp2);

    // neighbor means (bf16 gathers out of L2)
    sage_gather_mean<<<(NROOTS * 32) / 256, 256, 0, stream>>>(featb, forest1, x1b, NROOTS, 10);
    sage_gather_mean<<<(NH1    * 32) / 256, 256, 0, stream>>>(featb, forest2, x2b, NH1,    10);

    // layer 1: h1f = relu([F[forest1] | x2] @ W1)   (dominant GEMM, M=81920)
    sage_gemm_relu<<<NH1 / 16, 512, 0, stream>>>(featb, forest1, x2b, wp1, nullptr, h1fb, NH1);
    // layer 1 (roots): f0h = relu([F[forest0] | x1] @ W1)
    sage_gemm_relu<<<NROOTS / 16, 512, 0, stream>>>(featb, forest0, x1b, wp1, nullptr, f0hb, NROOTS);

    // mean over the 10 hop-1 children per root
    sage_group_mean10<<<(NROOTS * 32) / 256, 256, 0, stream>>>(h1fb, h1mb, NROOTS);

    // layer 2: out = relu([f0h | h1m] @ W2)  -> f32 output
    sage_gemm_relu<<<NROOTS / 16, 512, 0, stream>>>(f0hb, nullptr, h1mb, wp2, out, nullptr, NROOTS);
}
